// ISTFT_257698037845
// MI455X (gfx1250) — matrix operand order repaired and checked by reading, not compiled, not run on hardware
//
#include <hip/hip_runtime.h>

typedef __attribute__((ext_vector_type(16))) _Float16 v16h;
typedef __attribute__((ext_vector_type(8)))  _Float16 v8h;
typedef __attribute__((ext_vector_type(8)))  float    v8f;

#define WIN    1024
#define HOP    256
#define NBINS  513
#define TDIM   2048
#define BATCH  32
#define KTOT   1056          /* 2*513 = 1026 padded up to 33*32 */
#define NSTEPS 33            /* KTOT / 32 */
#define LOUT   (1 << 19)     /* 524288 output samples per batch */
#define PAD    384

union F16Frag { v16h v; v8h h[2]; };

// ---------------------------------------------------------------------------
// Kernel 1: windowed inverse-rDFT basis, f16, row-major [1024 x 1056]
//   cols 0..512   : win[w] * a_k/N * cos(2*pi*k*w/N)
//   cols 513..1025: win[w] * a_k/N * (-sin(2*pi*k*w/N))
//   cols 1026..   : zero (K padding)
// ---------------------------------------------------------------------------
__global__ void istft_initA(_Float16* __restrict__ Amat) {
  const int idx = blockIdx.x * 256 + threadIdx.x;   // 1024*1056 = 4224*256 exactly
  const int w   = idx / KTOT;
  const int kk  = idx - w * KTOT;
  const float c = 6.28318530717958647692f / 1024.f;
  const float win = 0.5f * (1.f - __cosf(c * (float)w));
  float val = 0.f;
  if (kk < 2 * NBINS) {
    const int   k     = (kk < NBINS) ? kk : kk - NBINS;
    const float scale = (k == 0 || k == 512) ? (1.f / 1024.f) : (2.f / 1024.f);
    const int   ph    = (k * w) & 1023;              // exact angle reduction
    const float ang   = c * (float)ph;
    val = win * scale * ((kk < NBINS) ? __cosf(ang) : -__sinf(ang));
  }
  Amat[idx] = (_Float16)val;
}

// ---------------------------------------------------------------------------
// Kernel 2: zero the output (harness poisons d_out; OLA atomics need zeros)
// ---------------------------------------------------------------------------
__global__ void istft_zero(float4* __restrict__ out) {
  out[(size_t)blockIdx.x * 256 + threadIdx.x] = make_float4(0.f, 0.f, 0.f, 0.f);
}

// ---------------------------------------------------------------------------
// Branch-free staging load: 8 f32 values, rows kbase..kbase+7 (wave-uniform),
// column tS. Re/Im select + padding zero-mask are scalar cndmasks; the row is
// clamped so padding rows do a harmless in-bounds load.
// ---------------------------------------------------------------------------
__device__ __forceinline__ void stage_load(const float* __restrict__ reB,
                                           const float* __restrict__ imB,
                                           int kbase, int tS, float* vals) {
#pragma unroll
  for (int j = 0; j < 8; ++j) {
    const int  kk    = kbase + j;          // wave-uniform (sgpr)
    const bool isRe  = kk < NBINS;
    const bool valid = kk < 2 * NBINS;
    int row = isRe ? kk : kk - NBINS;
    row = row > NBINS - 1 ? NBINS - 1 : row;
    const float* base = isRe ? reB : imB;  // scalar 64-bit select
    const float v = base[row * TDIM + tS];
    vals[j] = valid ? v : 0.f;
  }
}

// ---------------------------------------------------------------------------
// Kernel 3: WMMA GEMM  F = A(1024x1056,f16) * X(1056x65536,f16-staged)
//           with fused windowed overlap-add scatter into the output.
// Block = 256 threads (8 waves), tile 256(M) x 64(N); wave tile 64x32.
// Double-buffered LDS, one barrier per K-step, next-tile loads issued
// ahead of the current step's WMMAs.
// ---------------------------------------------------------------------------
__global__ __launch_bounds__(256)
void istft_gemm_ola(const float* __restrict__ re, const float* __restrict__ im,
                    const _Float16* __restrict__ Amat, float* __restrict__ out) {
  __shared__ _Float16 ldsB[2][64 * 40];  // 2 x (64 cols, K-stride 40 = 80B)

  const int tid   = threadIdx.x;
  const int lane  = tid & 31;
  const int wv    = tid >> 5;
  const int waveM = wv & 3;              // 4 waves along M
  const int waveN = wv >> 2;             // 2 waves along N

  const int c0      = blockIdx.x * 64;   // global column = b*T + t
  const int bIdx    = c0 >> 11;          // 64 | 2048 -> single batch per block
  const int t0      = c0 & (TDIM - 1);
  const int rowBase = blockIdx.y * 256 + waveM * 64;

  const float* reB = re + (size_t)bIdx * NBINS * TDIM;
  const float* imB = im + (size_t)bIdx * NBINS * TDIM;

  // staging coords: 256 threads cover 32 K-rows x 64 cols, 8 rows per thread;
  // tid>>6 is constant within a wave -> readfirstlane makes it provably sgpr.
  const int halfU = __builtin_amdgcn_readfirstlane(tid >> 6);  // 0..3
  const int colS  = tid & 63;
  const int tS    = t0 + colS;

  v8f acc[8] = {};                       // [i(M,4)][j(N,2)]

  const int kOffA = (lane >> 4) * 8;     // ISA 16-bit A layout: hi lanes K+8
  const int kOffB = (lane >> 4) * 8;     // B matches A: hi lanes K+8
  const int mLane = lane & 15;

  // ---- prologue: stage tile 0 into buffer 0 --------------------------------
  {
    float vals[8];
    stage_load(reB, imB, halfU * 8, tS, vals);
    v8h pack;
#pragma unroll
    for (int j = 0; j < 8; ++j) pack[j] = (_Float16)vals[j];
    *(v8h*)(&ldsB[0][colS * 40 + halfU * 8]) = pack;
  }

  for (int step = 0; step < NSTEPS; ++step) {
    const int kk0 = step * 32;
    const int buf = step & 1;
    __syncthreads();

    // ---- A fragments: two global_load_b128 per lane per M-tile (L2 hot) ----
    F16Frag afrag[4];
#pragma unroll
    for (int i = 0; i < 4; ++i) {
      const _Float16* ap =
          Amat + (size_t)(rowBase + i * 16 + mLane) * KTOT + kk0 + kOffA;
      afrag[i].h[0] = *(const v8h*)ap;
      afrag[i].h[1] = *(const v8h*)(ap + 16);
    }
    // ---- B fragments: two ds_load_b128 per lane per N-tile -----------------
    F16Frag bfrag[2];
#pragma unroll
    for (int j = 0; j < 2; ++j) {
      const _Float16* bp = &ldsB[buf][(waveN * 32 + j * 16 + mLane) * 40 + kOffB];
      bfrag[j].h[0] = *(const v8h*)bp;
      bfrag[j].h[1] = *(const v8h*)(bp + 16);
    }

    // ---- prefetch next X tile (global f32) before the matrix math ----------
    float vals[8];
    if (step < NSTEPS - 1)                       // scalar condition
      stage_load(reB, imB, kk0 + 32 + halfU * 8, tS, vals);

#pragma unroll
    for (int i = 0; i < 4; ++i)
#pragma unroll
      for (int j = 0; j < 2; ++j)
        acc[i * 2 + j] = __builtin_amdgcn_wmma_f32_16x16x32_f16(
            false, afrag[i].v, false, bfrag[j].v,
            (short)0, acc[i * 2 + j], false, false);

    // ---- convert + store next tile into the other buffer -------------------
    if (step < NSTEPS - 1) {
      v8h pack;
#pragma unroll
      for (int j = 0; j < 8; ++j) pack[j] = (_Float16)vals[j];
      *(v8h*)(&ldsB[buf ^ 1][colS * 40 + halfU * 8]) = pack;
    }
  }

  // ---- fused overlap-add: frames(w,t) scatter to y[t*HOP + w - PAD] --------
  // C/D layout: VGPR r, lanes 0-15 -> M=r, lanes 16-31 -> M=r+8; N = lane&15.
  const int wHalf = (lane >> 4) << 3;
  float* outB = out + ((size_t)bIdx << 19);
#pragma unroll
  for (int i = 0; i < 4; ++i) {
#pragma unroll
    for (int j = 0; j < 2; ++j) {
      const int c     = c0 + waveN * 32 + j * 16 + mLane;
      const int t     = c & (TDIM - 1);
      const int pBase = t * HOP + rowBase + i * 16 + wHalf - PAD;
#pragma unroll
      for (int r = 0; r < 8; ++r) {
        const int q = pBase + r;
        if (q >= 0 && q < LOUT) atomicAdd(outB + q, acc[i * 2 + j][r]);
      }
    }
  }
}

// ---------------------------------------------------------------------------
// Kernel 4: divide by the Hann^2 overlap envelope (edge-exact, <=4 terms)
// ---------------------------------------------------------------------------
__global__ void istft_norm(float* __restrict__ out) {
  const int q = blockIdx.x * 256 + threadIdx.x;   // LOUT/256 = 2048 blocks
  const int p = q + PAD;
  int tmax = p >> 8;            if (tmax > TDIM - 1) tmax = TDIM - 1;
  int tmin = (p - 768) >> 8;    if (tmin < 0) tmin = 0;  // ceil((p-1023)/256)
  const float c = 6.28318530717958647692f / 1024.f;
  float env = 0.f;
  for (int t = tmin; t <= tmax; ++t) {
    const int   w   = p - (t << 8);
    const float win = 0.5f * (1.f - __cosf(c * (float)w));
    env += win * win;
  }
  const float inv = 1.f / env;
#pragma unroll
  for (int b = 0; b < BATCH; ++b) out[((size_t)b << 19) + q] *= inv;
}

// ---------------------------------------------------------------------------
extern "C" void kernel_launch(void* const* d_in, const int* in_sizes, int n_in,
                              void* d_out, int out_size, void* d_ws, size_t ws_size,
                              hipStream_t stream) {
  const float* re   = (const float*)d_in[0];
  const float* im   = (const float*)d_in[1];
  float*       out  = (float*)d_out;
  _Float16*    Amat = (_Float16*)d_ws;   // 1024*1056*2 B ~= 2.06 MB workspace

  hipLaunchKernelGGL(istft_initA,    dim3(4224),    dim3(256), 0, stream, Amat);
  hipLaunchKernelGGL(istft_zero,     dim3(16384),   dim3(256), 0, stream, (float4*)d_out);
  hipLaunchKernelGGL(istft_gemm_ola, dim3(1024, 4), dim3(256), 0, stream, re, im, Amat, out);
  hipLaunchKernelGGL(istft_norm,     dim3(2048),    dim3(256), 0, stream, out);
}
